// MambaBlock_62345745269153
// MI455X (gfx1250) — compile-verified, matrix operand order checked by reading, not run on hardware
//
#include <hip/hip_runtime.h>
#include <hip/hip_bf16.h>

// ---------------------------------------------------------------------------
// Mamba block forward for gfx1250 (MI455X).
// BF16 WMMA GEMMs with async-to-LDS double-buffered B tiles + chunked scan.
// ---------------------------------------------------------------------------

typedef __attribute__((ext_vector_type(16))) __bf16 v16bf;
typedef __attribute__((ext_vector_type(8)))  __bf16 v8bf;
typedef __attribute__((ext_vector_type(8)))  float  v8f;

#define D_MODEL 1024
#define D_STATE 16
#define D_CONV  4
#define D_INNER 2048
#define DT_RANK 16
#define BATCH   2
#define SEQ     1024
#define MROWS   (BATCH * SEQ)            // 2048
#define XDBL_N  (DT_RANK + 2 * D_STATE)  // 48
#define NCHUNK  16
#define CHUNK   (SEQ / NCHUNK)           // 64

// ---------------------------------------------------------------------------
// CDNA5 async global->LDS copy (ASYNCcnt-tracked) + wait
// ---------------------------------------------------------------------------
__device__ __forceinline__ void async_copy_b128(unsigned lds_off, const void* g) {
    unsigned long long ga = (unsigned long long)g;
    asm volatile("global_load_async_to_lds_b128 %0, %1, off"
                 :: "v"(lds_off), "v"(ga) : "memory");
}
__device__ __forceinline__ void wait_async0() {
    asm volatile("s_wait_asynccnt 0x0" ::: "memory");
}

// ---------------------------------------------------------------------------
// fp32 -> bf16 conversion
// ---------------------------------------------------------------------------
__global__ void cvt_f32_bf16(const float* __restrict__ in,
                             __bf16* __restrict__ out, int n) {
    int i = blockIdx.x * blockDim.x + threadIdx.x;
    if (i < n) out[i] = (__bf16)in[i];
}

// ---------------------------------------------------------------------------
// WMMA fragment combine: elems 0..7 <- K=kb..kb+7, elems 8..15 <- K=kb+16..kb+23
// (CDNA5 ISA 7.12.2, 16-bit A 16x32; kb = lane<16 ? 0 : 8)
// ---------------------------------------------------------------------------
__device__ __forceinline__ v16bf frag_combine(v8bf lo, v8bf hi) {
    return __builtin_shufflevector(lo, hi, 0, 1, 2, 3, 4, 5, 6, 7,
                                           8, 9, 10, 11, 12, 13, 14, 15);
}
__device__ __forceinline__ v16bf load_frag(const __bf16* __restrict__ p) {
    return frag_combine(*(const v8bf*)(p), *(const v8bf*)(p + 16));
}

// ---------------------------------------------------------------------------
// LDS-staged GEMM: C = A(MxK bf16) @ Bw(NxK bf16)^T + bias, C fp32.
// Block = 256 threads = 8 waves; block computes a 128x64 macro-tile.
// Wave w -> rows [.. +16); all waves share a 64col x 32k B tile staged in LDS
// by global_load_async_to_lds_b128, double-buffered, one barrier per K-step.
// K must be a multiple of 32; M a multiple of 128; N a multiple of 64.
// ---------------------------------------------------------------------------
#define GEMM_COLS 64
#define LDS_COL_STRIDE 80   // bytes per column (32 bf16 + pad), 16B aligned
#define LDS_BUF_BYTES (GEMM_COLS * LDS_COL_STRIDE)

__global__ __launch_bounds__(256)
void gemm_bf16_wmma_lds(const __bf16* __restrict__ A,
                        const __bf16* __restrict__ Bw,
                        const float* __restrict__ bias,
                        float* __restrict__ C,
                        int M, int N, int K) {
    constexpr int TN = 4;
    __shared__ __align__(16) char smem[2 * LDS_BUF_BYTES];

    const int tid  = threadIdx.x;
    const int lane = tid & 31;
    const int wv   = tid >> 5;                  // 0..7
    const int colGroups = N / GEMM_COLS;
    const int rowGroup  = blockIdx.x / colGroups;
    const int cg        = blockIdx.x % colGroups;
    const int tileRow   = rowGroup * 8 + wv;
    const int nBase     = cg * GEMM_COLS;

    // A fragment source (direct global, register double-buffered)
    const int kb = (lane < 16) ? 0 : 8;
    const __bf16* aRow = A + (size_t)(tileRow * 16 + (lane & 15)) * K + kb;

    // cooperative B staging: thread loads 16B: col = tid/4, part = tid%4
    const int bcol  = tid >> 2;                 // 0..63
    const int bpart = tid & 3;                  // 0..3
    const __bf16* bSrc = Bw + (size_t)(nBase + bcol) * K + bpart * 8;
    const unsigned ldsBase =
        (unsigned)(size_t)(&smem[0]) + bcol * LDS_COL_STRIDE + bpart * 16;

    v8f acc[TN];
#pragma unroll
    for (int t = 0; t < TN; ++t)
        acc[t] = (v8f){0.f, 0.f, 0.f, 0.f, 0.f, 0.f, 0.f, 0.f};

    const int nsteps = K / 32;
    async_copy_b128(ldsBase, bSrc);             // k-step 0 -> buffer 0
    v16bf a_cur = load_frag(aRow);

    for (int i = 0; i < nsteps; ++i) {
        wait_async0();                           // my buf[i&1] load done
        __syncthreads();                         // everyone's load done,
                                                 // everyone's prev compute done
        if (i + 1 < nsteps)                      // overlap next tile with WMMAs
            async_copy_b128(ldsBase + ((i + 1) & 1) * LDS_BUF_BYTES,
                            bSrc + (i + 1) * 32);
        v16bf a_nxt = a_cur;
        if (i + 1 < nsteps) {
            __builtin_prefetch(aRow + (i + 2) * 32, 0, 3);
            a_nxt = load_frag(aRow + (i + 1) * 32);
        }

        const char* sb = (const char*)smem + (i & 1) * LDS_BUF_BYTES;
        v16bf b[TN];
#pragma unroll
        for (int t = 0; t < TN; ++t) {
            const char* colp =
                sb + (size_t)(t * 16 + (lane & 15)) * LDS_COL_STRIDE + kb * 2;
            b[t] = frag_combine(*(const v8bf*)(colp),
                                *(const v8bf*)(colp + 32));
        }
#pragma unroll
        for (int t = 0; t < TN; ++t)
            acc[t] = __builtin_amdgcn_wmma_f32_16x16x32_bf16(
                false, a_cur, false, b[t], (short)0, acc[t], false, false);
        a_cur = a_nxt;
    }

    const int rowOff = (lane < 16) ? 0 : 8;
#pragma unroll
    for (int t = 0; t < TN; ++t) {
        const int n = nBase + t * 16 + (lane & 15);
        const float bn = bias[n];
#pragma unroll
        for (int r = 0; r < 8; ++r) {
            const int m = tileRow * 16 + rowOff + r;
            C[(size_t)m * N + n] = acc[t][r] + bn;
        }
    }
}

// ---------------------------------------------------------------------------
// Direct-global WMMA GEMM for the skinny x_proj (N = 48).
// ---------------------------------------------------------------------------
template <int TN>
__global__ __launch_bounds__(256)
void gemm_bf16_wmma(const __bf16* __restrict__ A,
                    const __bf16* __restrict__ Bw,
                    const float* __restrict__ bias,
                    float* __restrict__ C,
                    int M, int N, int K) {
    const int wave = (blockIdx.x * blockDim.x + threadIdx.x) >> 5;
    const int lane = threadIdx.x & 31;
    const int colGroups = N / (16 * TN);
    const int tileRow = wave / colGroups;
    const int cg      = wave % colGroups;
    if (tileRow * 16 >= M) return;

    const int kb = (lane < 16) ? 0 : 8;
    const __bf16* aRow = A + (size_t)(tileRow * 16 + (lane & 15)) * K + kb;

    const int nBase = cg * (16 * TN);
    const __bf16* bRows[TN];
#pragma unroll
    for (int t = 0; t < TN; ++t)
        bRows[t] = Bw + (size_t)(nBase + t * 16 + (lane & 15)) * K + kb;

    v8f acc[TN];
#pragma unroll
    for (int t = 0; t < TN; ++t)
        acc[t] = (v8f){0.f, 0.f, 0.f, 0.f, 0.f, 0.f, 0.f, 0.f};

    // register-double-buffered so load waits overlap the WMMAs
    v16bf a_cur = load_frag(aRow);
    v16bf b_cur[TN];
#pragma unroll
    for (int t = 0; t < TN; ++t) b_cur[t] = load_frag(bRows[t]);

    for (int k0 = 0; k0 < K; k0 += 32) {
        v16bf a_nxt = a_cur;
        v16bf b_nxt[TN];
#pragma unroll
        for (int t = 0; t < TN; ++t) b_nxt[t] = b_cur[t];
        if (k0 + 32 < K) {
            a_nxt = load_frag(aRow + k0 + 32);
#pragma unroll
            for (int t = 0; t < TN; ++t)
                b_nxt[t] = load_frag(bRows[t] + k0 + 32);
        }
#pragma unroll
        for (int t = 0; t < TN; ++t)
            acc[t] = __builtin_amdgcn_wmma_f32_16x16x32_bf16(
                false, a_cur, false, b_cur[t], (short)0, acc[t], false, false);
        a_cur = a_nxt;
#pragma unroll
        for (int t = 0; t < TN; ++t) b_cur[t] = b_nxt[t];
    }

    const int rowOff = (lane < 16) ? 0 : 8;
#pragma unroll
    for (int t = 0; t < TN; ++t) {
        const int n  = nBase + t * 16 + (lane & 15);
        const float bn = bias[n];
#pragma unroll
        for (int r = 0; r < 8; ++r) {
            const int m = tileRow * 16 + rowOff + r;
            C[(size_t)m * N + n] = acc[t][r] + bn;
        }
    }
}

// ---------------------------------------------------------------------------
// Causal depthwise conv (k=4, left pad 3) + SiLU; fp32 + bf16 outputs.
// ---------------------------------------------------------------------------
__global__ void conv_silu_kernel(const float* __restrict__ x_and_res,
                                 const float* __restrict__ conv_w,
                                 const float* __restrict__ conv_b,
                                 float* __restrict__ xs_f,
                                 __bf16* __restrict__ xs_bf) {
    int i = blockIdx.x * blockDim.x + threadIdx.x;
    if (i >= MROWS * D_INNER) return;
    const int c  = i % D_INNER;
    const int bl = i / D_INNER;
    const int l  = bl % SEQ;
    const int b0 = bl - l;

    float acc = conv_b[c];
#pragma unroll
    for (int j = 0; j < D_CONV; ++j) {
        const int li = l - (D_CONV - 1) + j;
        if (li >= 0)
            acc += x_and_res[(size_t)(b0 + li) * (2 * D_INNER) + c] *
                   conv_w[c * D_CONV + j];
    }
    const float s = acc / (1.f + __expf(-acc));
    xs_f[i]  = s;
    xs_bf[i] = (__bf16)s;
}

// ---------------------------------------------------------------------------
// delta = softplus(dlt @ dt_proj_w^T + dt_proj_b)
// ---------------------------------------------------------------------------
__global__ void dt_softplus_kernel(const float* __restrict__ x_dbl,
                                   const float* __restrict__ dt_w,
                                   const float* __restrict__ dt_b,
                                   float* __restrict__ delta) {
    int i = blockIdx.x * blockDim.x + threadIdx.x;
    if (i >= MROWS * D_INNER) return;
    const int d  = i % D_INNER;
    const int bl = i / D_INNER;
    const float* row = x_dbl + (size_t)bl * XDBL_N;
    float acc = dt_b[d];
#pragma unroll
    for (int r = 0; r < DT_RANK; ++r)
        acc += row[r] * dt_w[d * DT_RANK + r];
    delta[i] = (acc > 20.f) ? acc : log1pf(__expf(acc));
}

// ---------------------------------------------------------------------------
// Chunked selective scan (h <- a*h + b is linear, so chunk-parallel):
//   pass1: per (b,d,chunk) compute per-state (prod a, partial sum)
//   pass2: per (b,d)      sequential prefix over 16 chunks -> chunk init states
//   pass3: per (b,d,chunk) replay chunk with true init state, emit gated bf16 y
// Thread mapping keeps d contiguous for coalescing.
// ---------------------------------------------------------------------------
__global__ __launch_bounds__(256)
void scan_pass1(const float* __restrict__ delta,
                const float* __restrict__ xs_f,
                const float* __restrict__ x_dbl,
                const float* __restrict__ A_log,
                float* __restrict__ Pbuf,   // [B][NCHUNK][16][D_INNER]
                float* __restrict__ Sbuf) {
    int t = blockIdx.x * blockDim.x + threadIdx.x;
    if (t >= BATCH * NCHUNK * D_INNER) return;
    const int d    = t % D_INNER;
    const int rest = t / D_INNER;
    const int c    = rest % NCHUNK;
    const int b    = rest / NCHUNK;

    float Arow[D_STATE], P[D_STATE], S[D_STATE];
#pragma unroll
    for (int n = 0; n < D_STATE; ++n) {
        Arow[n] = -__expf(A_log[d * D_STATE + n]);
        P[n] = 1.f;
        S[n] = 0.f;
    }
    const int l0 = c * CHUNK;
    for (int l = l0; l < l0 + CHUNK; ++l) {
        const size_t bl = (size_t)b * SEQ + l;
        const float dt = delta[bl * D_INNER + d];
        const float dx = dt * xs_f[bl * D_INNER + d];
        const float* dbl = x_dbl + bl * XDBL_N;
#pragma unroll
        for (int n = 0; n < D_STATE; ++n) {
            const float a = __expf(dt * Arow[n]);
            P[n] *= a;
            S[n] = a * S[n] + dx * dbl[DT_RANK + n];
        }
    }
#pragma unroll
    for (int n = 0; n < D_STATE; ++n) {
        const size_t o = (((size_t)b * NCHUNK + c) * D_STATE + n) * D_INNER + d;
        Pbuf[o] = P[n];
        Sbuf[o] = S[n];
    }
}

__global__ __launch_bounds__(256)
void scan_pass2(const float* __restrict__ Pbuf,
                const float* __restrict__ Sbuf,
                float* __restrict__ Hinit) {  // [B][NCHUNK][16][D_INNER]
    int t = blockIdx.x * blockDim.x + threadIdx.x;
    if (t >= BATCH * D_INNER) return;
    const int d = t % D_INNER;
    const int b = t / D_INNER;
    float h[D_STATE];
#pragma unroll
    for (int n = 0; n < D_STATE; ++n) h[n] = 0.f;
    for (int c = 0; c < NCHUNK; ++c) {
#pragma unroll
        for (int n = 0; n < D_STATE; ++n) {
            const size_t o = (((size_t)b * NCHUNK + c) * D_STATE + n) * D_INNER + d;
            Hinit[o] = h[n];
            h[n] = Pbuf[o] * h[n] + Sbuf[o];
        }
    }
}

__global__ __launch_bounds__(256)
void scan_pass3(const float* __restrict__ delta,
                const float* __restrict__ xs_f,
                const float* __restrict__ x_dbl,
                const float* __restrict__ x_and_res,
                const float* __restrict__ A_log,
                const float* __restrict__ Dp,
                const float* __restrict__ Hinit,
                __bf16* __restrict__ y_bf) {
    int t = blockIdx.x * blockDim.x + threadIdx.x;
    if (t >= BATCH * NCHUNK * D_INNER) return;
    const int d    = t % D_INNER;
    const int rest = t / D_INNER;
    const int c    = rest % NCHUNK;
    const int b    = rest / NCHUNK;

    float Arow[D_STATE], h[D_STATE];
#pragma unroll
    for (int n = 0; n < D_STATE; ++n) {
        Arow[n] = -__expf(A_log[d * D_STATE + n]);
        h[n] = Hinit[(((size_t)b * NCHUNK + c) * D_STATE + n) * D_INNER + d];
    }
    const float Dd = Dp[d];
    const int l0 = c * CHUNK;
    for (int l = l0; l < l0 + CHUNK; ++l) {
        const size_t bl = (size_t)b * SEQ + l;
        const float dt = delta[bl * D_INNER + d];
        const float xv = xs_f[bl * D_INNER + d];
        const float* dbl = x_dbl + bl * XDBL_N;
        const float dx = dt * xv;
        float y = 0.f;
#pragma unroll
        for (int n = 0; n < D_STATE; ++n) {
            const float hv = __expf(dt * Arow[n]) * h[n] + dx * dbl[DT_RANK + n];
            h[n] = hv;
            y += hv * dbl[DT_RANK + D_STATE + n];
        }
        y += xv * Dd;
        const float r = x_and_res[bl * (2 * D_INNER) + D_INNER + d];
        const float g = r / (1.f + __expf(-r));
        y_bf[bl * D_INNER + d] = (__bf16)(y * g);
    }
}

// ---------------------------------------------------------------------------
// Launch
// ---------------------------------------------------------------------------
extern "C" void kernel_launch(void* const* d_in, const int* in_sizes, int n_in,
                              void* d_out, int out_size, void* d_ws, size_t ws_size,
                              hipStream_t stream) {
    (void)in_sizes; (void)n_in; (void)out_size; (void)ws_size;

    const float* x          = (const float*)d_in[0];
    const float* in_proj_w  = (const float*)d_in[1];
    const float* in_proj_b  = (const float*)d_in[2];
    const float* conv_w     = (const float*)d_in[3];
    const float* conv_b     = (const float*)d_in[4];
    const float* x_proj_w   = (const float*)d_in[5];
    const float* x_proj_b   = (const float*)d_in[6];
    const float* dt_proj_w  = (const float*)d_in[7];
    const float* dt_proj_b  = (const float*)d_in[8];
    const float* A_log      = (const float*)d_in[9];
    const float* Dp         = (const float*)d_in[10];
    const float* out_proj_w = (const float*)d_in[11];
    const float* out_proj_b = (const float*)d_in[12];
    float* out = (float*)d_out;

    char* ws = (char*)d_ws;
    size_t off = 0;
    auto alloc = [&](size_t bytes) -> void* {
        void* p = ws + off;
        off = (off + bytes + 255) & ~(size_t)255;
        return p;
    };
    __bf16* x_bf      = (__bf16*)alloc((size_t)MROWS * D_MODEL * 2);
    __bf16* w_in_bf   = (__bf16*)alloc((size_t)2 * D_INNER * D_MODEL * 2);
    __bf16* w_xp_bf   = (__bf16*)alloc((size_t)XDBL_N * D_INNER * 2);
    __bf16* w_out_bf  = (__bf16*)alloc((size_t)D_MODEL * D_INNER * 2);
    float*  x_and_res = (float*)alloc((size_t)MROWS * 2 * D_INNER * 4);
    float*  xs_f      = (float*)alloc((size_t)MROWS * D_INNER * 4);
    __bf16* xs_bf     = (__bf16*)alloc((size_t)MROWS * D_INNER * 2);
    float*  x_dbl     = (float*)alloc((size_t)MROWS * XDBL_N * 4);
    float*  delta     = (float*)alloc((size_t)MROWS * D_INNER * 4);
    __bf16* y_bf      = (__bf16*)alloc((size_t)MROWS * D_INNER * 2);
    const size_t scanElems = (size_t)BATCH * NCHUNK * D_STATE * D_INNER;
    float*  Pbuf  = (float*)alloc(scanElems * 4);
    float*  Sbuf  = (float*)alloc(scanElems * 4);
    float*  Hinit = (float*)alloc(scanElems * 4);

    const int T = 256;
    auto blocks = [](size_t n, int t) { return (int)((n + t - 1) / t); };

    // 1. bf16 down-converts
    cvt_f32_bf16<<<blocks((size_t)MROWS * D_MODEL, T), T, 0, stream>>>(
        x, x_bf, MROWS * D_MODEL);
    cvt_f32_bf16<<<blocks((size_t)2 * D_INNER * D_MODEL, T), T, 0, stream>>>(
        in_proj_w, w_in_bf, 2 * D_INNER * D_MODEL);
    cvt_f32_bf16<<<blocks((size_t)XDBL_N * D_INNER, T), T, 0, stream>>>(
        x_proj_w, w_xp_bf, XDBL_N * D_INNER);
    cvt_f32_bf16<<<blocks((size_t)D_MODEL * D_INNER, T), T, 0, stream>>>(
        out_proj_w, w_out_bf, D_MODEL * D_INNER);

    // 2. in_proj GEMM: [2048,1024] x [4096,1024]^T -> [2048,4096]
    {
        int grid = (MROWS / 128) * ((2 * D_INNER) / GEMM_COLS);  // 16*64
        gemm_bf16_wmma_lds<<<grid, T, 0, stream>>>(
            x_bf, w_in_bf, in_proj_b, x_and_res, MROWS, 2 * D_INNER, D_MODEL);
    }

    // 3. causal depthwise conv + silu
    conv_silu_kernel<<<blocks((size_t)MROWS * D_INNER, T), T, 0, stream>>>(
        x_and_res, conv_w, conv_b, xs_f, xs_bf);

    // 4. x_proj GEMM: [2048,2048] x [48,2048]^T -> [2048,48]
    {
        int waves = (MROWS / 16) * (XDBL_N / 16);
        gemm_bf16_wmma<1><<<blocks((size_t)waves * 32, T), T, 0, stream>>>(
            xs_bf, w_xp_bf, x_proj_b, x_dbl, MROWS, XDBL_N, D_INNER);
    }

    // 5. dt_proj + softplus
    dt_softplus_kernel<<<blocks((size_t)MROWS * D_INNER, T), T, 0, stream>>>(
        x_dbl, dt_proj_w, dt_proj_b, delta);

    // 6. chunked selective scan (+ D skip + silu(res) gating) -> bf16
    scan_pass1<<<blocks((size_t)BATCH * NCHUNK * D_INNER, T), T, 0, stream>>>(
        delta, xs_f, x_dbl, A_log, Pbuf, Sbuf);
    scan_pass2<<<blocks((size_t)BATCH * D_INNER, T), T, 0, stream>>>(
        Pbuf, Sbuf, Hinit);
    scan_pass3<<<blocks((size_t)BATCH * NCHUNK * D_INNER, T), T, 0, stream>>>(
        delta, xs_f, x_dbl, x_and_res, A_log, Dp, Hinit, y_bf);

    // 7. out_proj GEMM: [2048,2048] x [1024,2048]^T -> [2048,1024]
    {
        int grid = (MROWS / 128) * (D_MODEL / GEMM_COLS);        // 16*16
        gemm_bf16_wmma_lds<<<grid, T, 0, stream>>>(
            y_bf, w_out_bf, out_proj_b, out, MROWS, D_MODEL, D_INNER);
    }
}
